// LSTM_266287972551
// MI455X (gfx1250) — matrix operand order repaired and checked by reading, not compiled, not run on hardware
//
#include <hip/hip_runtime.h>
#include <hip/hip_bf16.h>

// LSTM: 2 layers, H=64, B=512, T=3000, IN=21, FC(64->1).
// Fused persistent kernel: grid = 32 WGs (16 batch rows each), 256 threads (8 waves).
// Per step, per wave: 2 N-tiles x (3+4) K-chunks = 14 v_wmma_f32_16x16x32_bf16.
// Biases folded into the WMMA accumulator init; cell is branch-free v_exp/v_rcp.

#define HID   64
#define NG    256      // 4*H gate width
#define INF   21
#define TT    3000
#define BATCH 512
#define MTILE 16

typedef __bf16 v16bf __attribute__((ext_vector_type(16)));
typedef float  v8f   __attribute__((ext_vector_type(8)));

union BF16x16 { v16bf v; __bf16 e[16]; };
union F32x8   { v8f   v; float  e[8];  };

__device__ __forceinline__ v8f wmma_bf16(v16bf a, v16bf b, v8f c) {
    // (neg_a, A, neg_b, B, c_mod, C, reuse_a, reuse_b)
    return __builtin_amdgcn_wmma_f32_16x16x32_bf16(false, a, false, b, (short)0, c, false, false);
}

#define LOG2E   1.4426950408889634f
#define LOG2E2  2.8853900817779268f

// sigmoid(x) = 1/(1+e^-x): v_mul, v_exp_f32, v_add, v_rcp_f32 (branch-free)
__device__ __forceinline__ float sigm(float x) {
    return __builtin_amdgcn_rcpf(1.0f + __builtin_amdgcn_exp2f(x * -LOG2E));
}
// tanh(x) = 1 - 2/(e^{2x}+1): saturates correctly at +/-1 (branch-free)
__device__ __forceinline__ float ftanh(float x) {
    return 1.0f - 2.0f * __builtin_amdgcn_rcpf(1.0f + __builtin_amdgcn_exp2f(x * LOG2E2));
}

__global__ void __launch_bounds__(256, 1)
lstm_fused_kernel(const float* __restrict__ x,
                  const float* __restrict__ Wih0, const float* __restrict__ Whh0,
                  const float* __restrict__ bih0, const float* __restrict__ bhh0,
                  const float* __restrict__ Wih1, const float* __restrict__ Whh1,
                  const float* __restrict__ bih1, const float* __restrict__ bhh1,
                  const float* __restrict__ Wfc,  const float* __restrict__ bfc,
                  float* __restrict__ out)
{
    // LDS: gate exchange (16x256 f32), h fragments in WMMA A-layout, x frag (ping-pong), FC reduce.
    __shared__ float gatesS[MTILE * NG];                  // 16 KB
    __shared__ __align__(32) __bf16 h0frag[2 * 512];      // 2 KB (K-chunks 0..1 of h0)
    __shared__ __align__(32) __bf16 h1frag[2 * 512];      // 2 KB
    __shared__ __align__(32) __bf16 xfrag[2 * 512];       // 2 KB, double-buffered
    __shared__ float redS[256];                           // 1 KB

    const int tid  = threadIdx.x;
    const int lane = tid & 31;
    const int wave = tid >> 5;
    const int b0   = blockIdx.x * MTILE;

    // ---------------- one-time: pack weights into register B-fragments (bf16) ----
    // B[k][n] = W[n][k] (gates = act @ W^T). Wave owns N-tiles {2w, 2w+1}.
    // B-frag mapping: lane L -> N = nt*16 + (L&15); element e -> K = c*32 + (e>>3)*16 + (L>>4)*8 + (e&7).
    BF16x16 B0[2][3];   // layer0: K-chunks [x_pad(0..31) | Whh0(32..95)]
    BF16x16 B1[2][4];   // layer1: K-chunks [Wih1(0..63)  | Whh1(64..127)]
    F32x8   bias0v[2], bias1v[2];   // per-lane C-init: bias[N] broadcast over the 8 M-rows
    #pragma unroll
    for (int nt = 0; nt < 2; ++nt) {
        const int g = (2 * wave + nt) * 16 + (lane & 15);   // gate/output column
        #pragma unroll
        for (int c = 0; c < 3; ++c) {
            #pragma unroll
            for (int e = 0; e < 16; ++e) {
                const int K = c * 32 + ((e >> 3) * 16) + ((lane >> 4) * 8) + (e & 7);
                float v;
                if (K < 32) v = (K < INF) ? Wih0[g * INF + K] : 0.0f;
                else        v = Whh0[g * HID + (K - 32)];
                B0[nt][c].e[e] = (__bf16)v;
            }
        }
        #pragma unroll
        for (int c = 0; c < 4; ++c) {
            #pragma unroll
            for (int e = 0; e < 16; ++e) {
                const int K = c * 32 + ((e >> 3) * 16) + ((lane >> 4) * 8) + (e & 7);
                const float v = (K < HID) ? Wih1[g * HID + K] : Whh1[g * HID + (K - HID)];
                B1[nt][c].e[e] = (__bf16)v;
            }
        }
        const float bb0 = bih0[g] + bhh0[g];
        const float bb1 = bih1[g] + bhh1[g];
        #pragma unroll
        for (int rr = 0; rr < 8; ++rr) { bias0v[nt].e[rr] = bb0; bias1v[nt].e[rr] = bb1; }
    }

    // ---------------- per-thread cell assignment: row m, columns 4s..4s+3 ---------
    const int s = tid & 15;
    const int m = tid >> 4;
    float wfcv[4];
    int   haddr[4];          // A-fragment LDS slot for h[m][col]
    #pragma unroll
    for (int j = 0; j < 4; ++j) {
        const int col = 4 * s + j;
        wfcv[j] = Wfc[col];
        // A-layout inverse: L = m + 16*((K%16)/8); e = ((K%32)/16)*8 + K%8; chunk = K/32
        const int L  = m + 16 * ((col & 15) >> 3);
        const int ee = ((col & 31) >> 4) * 8 + (col & 7);
        haddr[j] = (col >> 5) * 512 + L * 16 + ee;
    }
    float c0s[4] = {0.f, 0.f, 0.f, 0.f};
    float c1s[4] = {0.f, 0.f, 0.f, 0.f};
    float hlast[4] = {0.f, 0.f, 0.f, 0.f};

    // x staging: 512 fragment elements/step, 2 per thread, in A-fragment order.
    int  xbase[2], xslot[2];
    bool xval[2];
    #pragma unroll
    for (int j2 = 0; j2 < 2; ++j2) {
        const int fe = 2 * tid + j2;
        const int L  = fe >> 4, e = fe & 15;
        const int Mx = L & 15;
        const int Kx = (e >> 3) * 16 + ((L >> 4) * 8) + (e & 7);
        xslot[j2] = L * 16 + e;
        xval[j2]  = (Kx < INF);
        xbase[j2] = (b0 + Mx) * (TT * INF) + Kx;
    }

    // zero initial hidden state fragments; stage x for t=0 into buffer 0
    for (int i = tid; i < 1024; i += 256) {
        h0frag[i] = (__bf16)0.0f;
        h1frag[i] = (__bf16)0.0f;
    }
    #pragma unroll
    for (int j2 = 0; j2 < 2; ++j2) {
        float v = 0.0f;
        if (xval[j2]) v = x[xbase[j2]];
        xfrag[xslot[j2]] = (__bf16)v;
    }
    __syncthreads();

    // ======================= sequential scan over time ===========================
    for (int t = 0; t < TT; ++t) {
        const int pc = (t & 1) * 512;         // current x buffer
        const int pn = ((t + 1) & 1) * 512;   // next x buffer

        // -- layer0 gates: [x_pad(16x32) | h0(16x64)] @ B0 (96x256) + bias0 --
        {
            const v16bf a0 = *(const v16bf*)&xfrag[pc + lane * 16];
            const v16bf a1 = *(const v16bf*)&h0frag[lane * 16];
            const v16bf a2 = *(const v16bf*)&h0frag[512 + lane * 16];
            #pragma unroll
            for (int nt = 0; nt < 2; ++nt) {
                v8f acc = bias0v[nt].v;
                acc = wmma_bf16(a0, B0[nt][0].v, acc);
                acc = wmma_bf16(a1, B0[nt][1].v, acc);
                acc = wmma_bf16(a2, B0[nt][2].v, acc);
                F32x8 r; r.v = acc;
                const int N  = (2 * wave + nt) * 16 + (lane & 15);
                const int Mb = 8 * (lane >> 4);
                #pragma unroll
                for (int rr = 0; rr < 8; ++rr)
                    gatesS[(Mb + rr) * NG + N] = r.e[rr];
            }
        }
        __syncthreads();

        // -- layer0 cell + stage x(t+1) (load latency hides behind mm1/cell1) --
        if (t + 1 < TT) {
            #pragma unroll
            for (int j2 = 0; j2 < 2; ++j2) {
                float v = 0.0f;
                if (xval[j2]) {
                    v = x[xbase[j2] + (t + 1) * INF];
                    __builtin_prefetch(&x[xbase[j2] + (t + 2) * INF], 0, 0);
                }
                xfrag[pn + xslot[j2]] = (__bf16)v;
            }
        }
        #pragma unroll
        for (int j = 0; j < 4; ++j) {
            const int col = 4 * s + j;
            const float gi = sigm (gatesS[m * NG + col]);
            const float gf = sigm (gatesS[m * NG + 64  + col]);
            const float gg = ftanh(gatesS[m * NG + 128 + col]);
            const float go = sigm (gatesS[m * NG + 192 + col]);
            const float cc = gf * c0s[j] + gi * gg;
            c0s[j] = cc;
            h0frag[haddr[j]] = (__bf16)(go * ftanh(cc));
        }
        __syncthreads();

        // -- layer1 gates: [h0(16x64) | h1(16x64)] @ B1 (128x256) + bias1 --
        {
            const v16bf a0 = *(const v16bf*)&h0frag[lane * 16];
            const v16bf a1 = *(const v16bf*)&h0frag[512 + lane * 16];
            const v16bf a2 = *(const v16bf*)&h1frag[lane * 16];
            const v16bf a3 = *(const v16bf*)&h1frag[512 + lane * 16];
            #pragma unroll
            for (int nt = 0; nt < 2; ++nt) {
                v8f acc = bias1v[nt].v;
                acc = wmma_bf16(a0, B1[nt][0].v, acc);
                acc = wmma_bf16(a1, B1[nt][1].v, acc);
                acc = wmma_bf16(a2, B1[nt][2].v, acc);
                acc = wmma_bf16(a3, B1[nt][3].v, acc);
                F32x8 r; r.v = acc;
                const int N  = (2 * wave + nt) * 16 + (lane & 15);
                const int Mb = 8 * (lane >> 4);
                #pragma unroll
                for (int rr = 0; rr < 8; ++rr)
                    gatesS[(Mb + rr) * NG + N] = r.e[rr];
            }
        }
        __syncthreads();

        // -- layer1 cell --
        #pragma unroll
        for (int j = 0; j < 4; ++j) {
            const int col = 4 * s + j;
            const float gi = sigm (gatesS[m * NG + col]);
            const float gf = sigm (gatesS[m * NG + 64  + col]);
            const float gg = ftanh(gatesS[m * NG + 128 + col]);
            const float go = sigm (gatesS[m * NG + 192 + col]);
            const float cc = gf * c1s[j] + gi * gg;
            c1s[j] = cc;
            const float hh = go * ftanh(cc);
            hlast[j] = hh;
            h1frag[haddr[j]] = (__bf16)hh;
        }
        __syncthreads();
    }

    // ======================= final FC: out[b] = h1_last . Wfc + bfc ==============
    float p = hlast[0] * wfcv[0] + hlast[1] * wfcv[1]
            + hlast[2] * wfcv[2] + hlast[3] * wfcv[3];
    redS[m * 16 + s] = p;
    __syncthreads();
    if (tid < 16) {
        float acc = bfc[0];
        #pragma unroll
        for (int k = 0; k < 16; ++k) acc += redS[tid * 16 + k];
        out[b0 + tid] = acc;
    }
}

extern "C" void kernel_launch(void* const* d_in, const int* in_sizes, int n_in,
                              void* d_out, int out_size, void* d_ws, size_t ws_size,
                              hipStream_t stream) {
    (void)in_sizes; (void)n_in; (void)d_ws; (void)ws_size; (void)out_size;
    const float* x    = (const float*)d_in[0];
    const float* Wih0 = (const float*)d_in[1];
    const float* Whh0 = (const float*)d_in[2];
    const float* bih0 = (const float*)d_in[3];
    const float* bhh0 = (const float*)d_in[4];
    const float* Wih1 = (const float*)d_in[5];
    const float* Whh1 = (const float*)d_in[6];
    const float* bih1 = (const float*)d_in[7];
    const float* bhh1 = (const float*)d_in[8];
    const float* Wfc  = (const float*)d_in[9];
    const float* bfc  = (const float*)d_in[10];
    float* out = (float*)d_out;

    lstm_fused_kernel<<<dim3(BATCH / MTILE), dim3(256), 0, stream>>>(
        x, Wih0, Whh0, bih0, bhh0, Wih1, Whh1, bih1, bhh1, Wfc, bfc, out);
}
